// NTXentLoss_66202625901204
// MI455X (gfx1250) — matrix operand order repaired and checked by reading, not compiled, not run on hardware
//
#include <hip/hip_runtime.h>
#include <hip/hip_bf16.h>

typedef __attribute__((ext_vector_type(16))) _Float16 v16h;
typedef __attribute__((ext_vector_type(8)))  float    v8f;

#define B_ROWS 4096
#define N_ROWS 8192
#define DDIM   256
#define LDSB   264   // padded LDS stride in halfs (16B-aligned, bank-spread)
#define CT_STEP 32   // columns staged per block iteration (2 x 16-wide tiles)

#define C_SCALE 2.8853900817779268f   // (1/TEMPERATURE) * log2(e) = 2*log2(e)

union HalfPack {
    v16h  v;
    uint4 q[2];
};

// ---------------------------------------------------------------------------
// Kernel 1: L2-normalize each row of Z = concat(z_i, z_j), store as f16.
// One 256-thread block per row (D = 256, one element per thread).
// ---------------------------------------------------------------------------
__global__ void __launch_bounds__(256)
ntx_normalize(const float* __restrict__ zi,
              const float* __restrict__ zj,
              _Float16* __restrict__ Zh) {
    const int row = blockIdx.x;
    const int t   = threadIdx.x;
    const float* src = (row < B_ROWS) ? (zi + (size_t)row * DDIM)
                                      : (zj + (size_t)(row - B_ROWS) * DDIM);
    const float v = src[t];
    float ss = v * v;
    #pragma unroll
    for (int m = 16; m >= 1; m >>= 1) ss += __shfl_xor(ss, m, 32);

    __shared__ float wsum[8];
    __shared__ float inv_s;
    if ((t & 31) == 0) wsum[t >> 5] = ss;
    __syncthreads();
    if (t == 0) {
        float s = 0.f;
        #pragma unroll
        for (int i = 0; i < 8; ++i) s += wsum[i];
        inv_s = 1.0f / fmaxf(sqrtf(s), 1e-12f);   // torch F.normalize eps clamp
    }
    __syncthreads();
    Zh[(size_t)row * DDIM + t] = (_Float16)(v * inv_s);
}

// ---------------------------------------------------------------------------
// Kernel 2: fused sim GEMM (WMMA f16->f32) + exp + per-row sum.
// Grid: 64 blocks x 256 threads. Block owns 128 rows; wave w owns a 16-row
// tile with its full A panel (16x256 f16) resident in VGPRs. Block loops over
// 256 iterations of 32 columns, staging each 32x256 B panel once in LDS.
// The diagonal is deliberately INCLUDED here (no branches); kernel 3 removes
// it analytically via rowsum[i] - exp(sim_ii).
// ---------------------------------------------------------------------------
__global__ void __launch_bounds__(256)
ntx_sim_rowsum(const _Float16* __restrict__ Zh,
               float* __restrict__ rowsum) {
    __shared__ _Float16 Bs[CT_STEP * LDSB];   // 16.9 KB

    const int t    = threadIdx.x;
    const int lane = t & 31;
    const int wv   = t >> 5;
    const int hi   = lane >> 4;     // half-wave select (0/1)
    const int l15  = lane & 15;
    const int row0 = blockIdx.x * 128 + wv * 16;

    // Preload this wave's A panel (16 rows x K=256) into registers.
    // A k order used here (same as B): lane L holds row L%16; lanes 0-15 carry K
    // [s*32, +16); lanes 16-31 K=[s*32+16, +16), contiguous per half.
    v16h afrag[8];
    {
        const _Float16* Arow = Zh + (size_t)(row0 + l15) * DDIM;
        #pragma unroll
        for (int s = 0; s < 8; ++s) {
            HalfPack u;
            const int kb = s * 32 + hi * 16;
            u.q[0] = *(const uint4*)(Arow + kb);
            u.q[1] = *(const uint4*)(Arow + kb + 8);
            afrag[s] = u.v;
        }
    }

    float rsum[8];
    #pragma unroll
    for (int v = 0; v < 8; ++v) rsum[v] = 0.f;

    const int bcol = t >> 3;          // staging column (0..31)
    const int bks  = (t & 7) * 32;    // staging k-offset (32 halfs per thread)

    for (int ct = 0; ct < N_ROWS / CT_STEP; ++ct) {
        const int c0 = ct * CT_STEP;
        __syncthreads();              // prior iteration done with Bs
        {
            const _Float16* src = Zh + (size_t)(c0 + bcol) * DDIM + bks;
            _Float16* dst = &Bs[bcol * LDSB + bks];
            #pragma unroll
            for (int q = 0; q < 4; ++q)
                *(uint4*)(dst + q * 8) = *(const uint4*)(src + q * 8);
            if (ct + 1 < N_ROWS / CT_STEP) {   // prefetch next B panel
                __builtin_prefetch(src + CT_STEP * DDIM, 0, 3);
            }
        }
        __syncthreads();

        #pragma unroll
        for (int sub = 0; sub < 2; ++sub) {
            // B k order used here (same as A): lane n holds column n; lanes 0-15 K=[s*32,+16),
            // lanes 16-31 K=[s*32+16,+16), contiguous per half.
            HalfPack bf[8];
            const _Float16* bbase = &Bs[(sub * 16 + l15) * LDSB + hi * 16];
            #pragma unroll
            for (int s = 0; s < 8; ++s) {
                bf[s].q[0] = *(const uint4*)(bbase + s * 32);
                bf[s].q[1] = *(const uint4*)(bbase + s * 32 + 8);
            }
            v8f acc = {};
            #pragma unroll
            for (int s = 0; s < 8; ++s) {
                acc = __builtin_amdgcn_wmma_f32_16x16x32_f16(
                    /*neg_a=*/false, afrag[s], /*neg_b=*/false, bf[s].v,
                    /*c_mod=*/(short)0, acc,
                    /*reuse_a=*/false, /*reuse_b=*/false);
            }
            // exp(sim) = exp2(acc * 2*log2e); raw v_exp_f32 (args bounded,
            // no OCML range fix-up needed). Branch-free.
            #pragma unroll
            for (int v = 0; v < 8; ++v)
                rsum[v] += __builtin_amdgcn_exp2f(acc[v] * C_SCALE);
        }
    }

    // Reduce row sums across the 16-lane N dimension of each half-wave.
    // C/D layout: VGPR v holds rows v / v+8 on lanes 0-15 / 16-31.
    #pragma unroll
    for (int v = 0; v < 8; ++v) {
        float r = rsum[v];
        r += __shfl_xor(r, 1, 32);
        r += __shfl_xor(r, 2, 32);
        r += __shfl_xor(r, 4, 32);
        r += __shfl_xor(r, 8, 32);
        if (l15 == 0) rowsum[row0 + v + (hi << 3)] = r;   // exclusive owner
    }
}

// ---------------------------------------------------------------------------
// Kernel 3: per-row loss term. One wave per row (8 rows / 256-thread block).
//   self = <z_i, z_i>  (f16 data, f32 accumulate)  -> diag logit = 2*self
//   pair = <z_i, z_{i^B}>                          -> positive logit = 2*pair
//   term[i] = 2*pair - log(rowsum[i] - exp(2*self))
// ---------------------------------------------------------------------------
__global__ void __launch_bounds__(256)
ntx_rowterm(const _Float16* __restrict__ Zh,
            const float* __restrict__ rowsum,
            float* __restrict__ term) {
    const int lane = threadIdx.x & 31;
    const int wv   = threadIdx.x >> 5;
    const int row  = blockIdx.x * 8 + wv;
    const int prow = row ^ B_ROWS;

    union { uint4 q; _Float16 h[8]; } ua, ub;
    ua.q = *(const uint4*)(Zh + (size_t)row  * DDIM + lane * 8);
    ub.q = *(const uint4*)(Zh + (size_t)prow * DDIM + lane * 8);

    float selfs = 0.f, pairs = 0.f;
    #pragma unroll
    for (int k = 0; k < 8; ++k) {
        const float av = (float)ua.h[k];
        const float bv = (float)ub.h[k];
        selfs += av * av;
        pairs += av * bv;
    }
    #pragma unroll
    for (int m = 16; m >= 1; m >>= 1) {
        selfs += __shfl_xor(selfs, m, 32);
        pairs += __shfl_xor(pairs, m, 32);
    }
    if (lane == 0) {
        const float eself = __builtin_amdgcn_exp2f(selfs * C_SCALE);
        term[row] = 2.0f * pairs - logf(rowsum[row] - eself);
    }
}

// ---------------------------------------------------------------------------
// Kernel 4: loss = -(1/N) * sum_i term[i]
// ---------------------------------------------------------------------------
__global__ void __launch_bounds__(256)
ntx_finalize(const float* __restrict__ term,
             float* __restrict__ out) {
    const int t = threadIdx.x;
    float acc = 0.f;
    for (int i = t; i < N_ROWS; i += 256) acc += term[i];
    #pragma unroll
    for (int m = 16; m >= 1; m >>= 1) acc += __shfl_xor(acc, m, 32);
    __shared__ float wsum[8];
    if ((t & 31) == 0) wsum[t >> 5] = acc;
    __syncthreads();
    if (t == 0) {
        float s = 0.f;
        #pragma unroll
        for (int i = 0; i < 8; ++i) s += wsum[i];
        out[0] = -s / (float)N_ROWS;
    }
}

extern "C" void kernel_launch(void* const* d_in, const int* in_sizes, int n_in,
                              void* d_out, int out_size, void* d_ws, size_t ws_size,
                              hipStream_t stream) {
    (void)in_sizes; (void)n_in; (void)out_size; (void)ws_size;
    const float* zi = (const float*)d_in[0];
    const float* zj = (const float*)d_in[1];

    _Float16* Zh   = (_Float16*)d_ws;                               // 4 MB
    float* rowsum  = (float*)((char*)d_ws +
                              (size_t)N_ROWS * DDIM * sizeof(_Float16));
    float* term    = rowsum + N_ROWS;                               // +32KB each

    ntx_normalize <<<N_ROWS,     256, 0, stream>>>(zi, zj, Zh);
    ntx_sim_rowsum<<<64,         256, 0, stream>>>(Zh, rowsum);
    ntx_rowterm   <<<N_ROWS / 8, 256, 0, stream>>>(Zh, rowsum, term);
    ntx_finalize  <<<1,          256, 0, stream>>>(term, (float*)d_out);
}
